// SegmentationMapPredictor_21208548508353
// MI455X (gfx1250) — matrix operand order repaired and checked by reading, not compiled, not run on hardware
//
#include <hip/hip_runtime.h>
#include <math.h>

typedef __attribute__((ext_vector_type(16))) __bf16       v16bf;
typedef __attribute__((ext_vector_type(8)))  float        v8f;
typedef __attribute__((ext_vector_type(4)))  unsigned int v4u;
typedef __attribute__((ext_vector_type(8)))  int          v8i;
typedef __attribute__((ext_vector_type(4)))  int          v4i;

union pk2   { __bf16 h[2];       unsigned int u; };
union frag16 { unsigned int u[8]; v16bf v; };

// ---------------------------------------------------------------------------
// A-matrix 16x32 (MxK) bf16 fragment from fp32 global (converted in-flight).
//   lane L: M = L%16, half = L/16 ; vgpr j: K = (j%4)*2 + (j/4)*16 + half*8
// ---------------------------------------------------------------------------
__device__ __forceinline__ v16bf load_a_frag_f32(const float* __restrict__ X, int ld,
                                                 int mbase, int kbase, int lane) {
    int m    = lane & 15;
    int half = lane >> 4;
    const float* row = X + (size_t)(mbase + m) * ld + kbase;
    v16bf a;
#pragma unroll
    for (int j = 0; j < 8; ++j) {
        int k0 = ((j & 3) << 1) + ((j >> 2) << 4) + (half << 3);
        a[2 * j]     = (__bf16)row[k0];
        a[2 * j + 1] = (__bf16)row[k0 + 1];
    }
    return a;
}

// A fragment from TDM-staged fp32 LDS tile (row stride `ld` floats, padded).
__device__ __forceinline__ v16bf load_a_frag_lds_f32(const float* lds, int ld,
                                                     int kbase, int lane) {
    int m    = lane & 15;
    int half = lane >> 4;
    const float* row = lds + m * ld + kbase;
    v16bf a;
#pragma unroll
    for (int j = 0; j < 8; ++j) {
        int k0 = ((j & 3) << 1) + ((j >> 2) << 4) + (half << 3);
        a[2 * j]     = (__bf16)row[k0];
        a[2 * j + 1] = (__bf16)row[k0 + 1];
    }
    return a;
}

// ---------------------------------------------------------------------------
// B-matrix 32x16 (KxN) bf16 from fp32 global: B[k][n] = Y[nbase+n][kbase+k].
//   lane L: N = L%16, khalf = L/16 ; vgpr j: K = (j%4)*2 + (j/4)*16 + khalf*8
// ---------------------------------------------------------------------------
__device__ __forceinline__ v16bf load_b_frag_f32(const float* __restrict__ Y, int ld,
                                                 int nbase, int kbase, int lane, int nmax) {
    int n     = lane & 15;
    int khalf = lane >> 4;
    int r = nbase + n;
    if (r >= nmax) r = nmax - 1;
    const float* row = Y + (size_t)r * ld + kbase;
    v16bf b;
#pragma unroll
    for (int j = 0; j < 8; ++j) {
        int k0 = ((j & 3) << 1) + ((j >> 2) << 4) + (khalf << 3);
        b[2 * j]     = (__bf16)row[k0];
        b[2 * j + 1] = (__bf16)row[k0 + 1];
    }
    return b;
}

// B fragment from pre-packed bf16 rows: two runs of 4 consecutive b32 loads (merge to two b128).
__device__ __forceinline__ v16bf load_b_frag_bf16(const unsigned short* __restrict__ Y,
                                                  int ld, int nbase, int kbase,
                                                  int lane, int nmax) {
    int n     = lane & 15;
    int khalf = lane >> 4;
    int r = nbase + n;
    if (r >= nmax) r = nmax - 1;
    const unsigned int* p =
        (const unsigned int*)(Y + (size_t)r * ld + kbase);   // pair k0/2 = (j%4) + (j/4)*8 + khalf*4
    frag16 x;
#pragma unroll
    for (int j = 0; j < 8; ++j) x.u[j] = p[(j & 3) + ((j >> 2) << 3) + (khalf << 2)];
    return x.v;
}

// ---------------------------------------------------------------------------
// One dense layer: Y = act(X @ W^T + bias),  X:[M,D], W:[D,D] row-major.
// ---------------------------------------------------------------------------
__global__ void mlp_layer_kernel(const float* __restrict__ X,
                                 const float* __restrict__ W,
                                 const float* __restrict__ bias,
                                 float* __restrict__ Y,
                                 int D, int relu) {
    int lane  = threadIdx.x & 31;
    int wid   = threadIdx.x >> 5;
    int mbase = blockIdx.x * 16;
    int nbase = (blockIdx.y * 8 + wid) * 16;
    if (nbase >= D) return;

    v8f c = {};
    for (int kb = 0; kb < D; kb += 32) {
        v16bf a = load_a_frag_f32(X, D, mbase, kb, lane);
        v16bf b = load_b_frag_f32(W, D, nbase, kb, lane, 0x40000000);
        c = __builtin_amdgcn_wmma_f32_16x16x32_bf16(false, a, false, b,
                                                    (short)0, c, false, false);
    }

    int n    = lane & 15;
    int half = lane >> 4;
    float bv = bias[nbase + n];
#pragma unroll
    for (int j = 0; j < 8; ++j) {
        int m   = mbase + j + (half << 3);           // C layout: M = vgpr + 8*half
        float v = c[j] + bv;
        if (relu) v = fmaxf(v, 0.0f);
        Y[(size_t)m * D + nbase + n] = v;
    }
}

// fp32 -> packed bf16 pairs (row-major preserved); npairs = total_elems/2.
__global__ void pack_bf16_kernel(const float* __restrict__ X,
                                 unsigned short* __restrict__ Ybf, int npairs) {
    int i = blockIdx.x * 256 + threadIdx.x;
    if (i >= npairs) return;
    pk2 t;
    t.h[0] = (__bf16)X[2 * i];
    t.h[1] = (__bf16)X[2 * i + 1];
    ((unsigned int*)Ybf)[i] = t.u;
}

// ---------------------------------------------------------------------------
// logits[b,f,q] = sum_d fvals[b,f,d] * qbf[b,q,d], scatter-added into
// out[b, h(f), w(f), q].
//   * 16 x D fp32 feature tile DMA'd into LDS by the Tensor Data Mover
//     (single tensor_load_to_lds, TENSORcnt-tracked, TDM row padding of
//      5 DWORDs -> LDS row stride 261 floats, odd stride = conflict-free
//      16-row fragment reads)
//   * A fragments resident in registers across all query tiles
//   * B fragments are packed-bf16 b128 loads from L2-resident qbf
// ---------------------------------------------------------------------------
#define LDS_LDF 261   // 256 floats + 5-DWORD TDM pad per row

__global__ void logits_scatter_kernel(const unsigned short* __restrict__ qbf, // [B*Q, D] bf16
                                      const float* __restrict__ fvals,        // [B*F, D] f32
                                      const int*   __restrict__ fidx,         // [B*F, 3]
                                      const int*   __restrict__ hptr,
                                      const int*   __restrict__ wptr,
                                      float* __restrict__ out,                // [B,H,W,Q]
                                      int Q, int F, int D) {
    __shared__ float smem[16 * LDS_LDF];

    int lane   = threadIdx.x & 31;
    int wid    = threadIdx.x >> 5;
    int ftiles = F >> 4;
    int b      = blockIdx.x / ftiles;
    int ft     = blockIdx.x - b * ftiles;
    int fbase  = b * F + ft * 16;               // global feature row base
    int H  = *hptr;
    int Wd = *wptr;

    // ---- TDM: DMA the 16 x D fp32 tile into LDS (wave 0 issues) ----
    if (wid == 0) {
        unsigned long long gaddr = (unsigned long long)(fvals + (size_t)fbase * D);
        unsigned int lds_off = (unsigned int)(size_t)&smem[0];   // addr[31:0] = LDS offset

        v4u g0;
        g0[0] = 1u;                                             // count=1 valid descriptor
        g0[1] = lds_off;                                        // D#.lds_addr
        g0[2] = (unsigned int)gaddr;                            // global_addr[31:0]
        g0[3] = (unsigned int)((gaddr >> 32) & 0x01FFFFFFu)     // global_addr[56:32]
              | (2u << 30);                                     // type = 2 ("image")

        v8i g1;
        g1[0] = (int)((2u << 16)      // data_size = 4 bytes
                    | (1u << 20)      // pad_enable
                    | (7u << 22)      // pad_interval: after 256 DWORDs (one row)
                    | (4u << 25));    // pad_amount: 5 DWORDs
        g1[1] = (int)((D & 0xFFFF) << 16);                      // tensor_dim0[15:0]
        g1[2] = (int)(((unsigned)D >> 16) | (16u << 16));       // dim0 hi | tensor_dim1=16
        g1[3] = (int)((D & 0xFFFF) << 16);                      // tile_dim0 = D
        g1[4] = 16;                                             // tile_dim1=16, tile_dim2=0
        g1[5] = D;                                              // tensor_dim0_stride lo
        g1[6] = 0;
        g1[7] = 0;

        v4i gz4 = {0, 0, 0, 0};                                 // groups 2/3: 2-D tensor
        v8i gz8 = {0, 0, 0, 0, 0, 0, 0, 0};
        __builtin_amdgcn_tensor_load_to_lds(g0, g1, gz4, gz4, gz8, 0);
        __builtin_amdgcn_s_wait_tensorcnt(0);
    }
    __syncthreads();

    // ---- resident A fragments over full K (D/32 chunks) ----
    v16bf afrag[8];
#pragma unroll
    for (int kc = 0; kc < 8; ++kc)
        afrag[kc] = load_a_frag_lds_f32(smem, LDS_LDF, kc * 32, lane);

    int qtiles = (Q + 15) >> 4;
    int n      = lane & 15;
    int half   = lane >> 4;
    int qrow0  = b * Q;

    for (int qt = wid; qt < qtiles; qt += 8) {
        int qbase = qt * 16;
        v8f c = {};
#pragma unroll
        for (int kc = 0; kc < 8; ++kc) {
            v16bf bf = load_b_frag_bf16(qbf, D, qrow0 + qbase, kc * 32, lane, qrow0 + Q);
            c = __builtin_amdgcn_wmma_f32_16x16x32_bf16(false, afrag[kc], false, bf,
                                                        (short)0, c, false, false);
        }

        int qidx = qbase + n;
        if (qidx < Q) {                           // qvalid mask (uniform offsets)
#pragma unroll
            for (int j = 0; j < 8; ++j) {
                int frow = fbase + j + (half << 3);    // C layout: M = vgpr + 8*half
                int fb = fidx[frow * 3 + 0];
                int fh = fidx[frow * 3 + 1];
                int fw = fidx[frow * 3 + 2];
                size_t off = ((((size_t)fb * H + fh) * Wd + fw) * (size_t)Q) + qidx;
                atomicAdd(&out[off], c[j]);
            }
        }
    }
}

// ---------------------------------------------------------------------------
extern "C" void kernel_launch(void* const* d_in, const int* in_sizes, int n_in,
                              void* d_out, int out_size, void* d_ws, size_t ws_size,
                              hipStream_t stream) {
    const float* queries = (const float*)d_in[0];
    const float* fvals   = (const float*)d_in[1];
    const int*   fidx    = (const int*)d_in[2];
    const int*   hptr    = (const int*)d_in[5];
    const int*   wptr    = (const int*)d_in[6];
    const float* Wl[4] = { (const float*)d_in[7],  (const float*)d_in[9],
                           (const float*)d_in[11], (const float*)d_in[13] };
    const float* bl[4] = { (const float*)d_in[8],  (const float*)d_in[10],
                           (const float*)d_in[12], (const float*)d_in[14] };

    int B = in_sizes[3] - 1;                                // offsets has B+1 entries
    int D = (int)lround(sqrt((double)in_sizes[7]));         // W0 is D x D
    int Q = in_sizes[0] / (B * D);                          // uniform batches
    int F = in_sizes[1] / (B * D);

    size_t qElems = (size_t)B * Q * D;
    float*          t0  = (float*)d_ws;
    float*          t1  = t0 + qElems;
    unsigned short* qbf = (unsigned short*)(t1 + qElems);

    // Output is accumulate-only -> zero it (capture-safe async memset).
    (void)hipMemsetAsync(d_out, 0, (size_t)out_size * sizeof(float), stream);

    dim3 mlpGrid((B * Q) / 16, (D / 16 + 7) / 8);
    mlp_layer_kernel<<<mlpGrid, 256, 0, stream>>>(queries, Wl[0], bl[0], t0, D, 1);
    mlp_layer_kernel<<<mlpGrid, 256, 0, stream>>>(t0,      Wl[1], bl[1], t1, D, 1);
    mlp_layer_kernel<<<mlpGrid, 256, 0, stream>>>(t1,      Wl[2], bl[2], t0, D, 1);
    mlp_layer_kernel<<<mlpGrid, 256, 0, stream>>>(t0,      Wl[3], bl[3], t1, D, 0);

    int npairs = (int)(qElems / 2);
    pack_bf16_kernel<<<(npairs + 255) / 256, 256, 0, stream>>>(t1, qbf, npairs);

    dim3 lsGrid(B * (F / 16));
    logits_scatter_kernel<<<lsGrid, 256, 0, stream>>>(qbf, fvals, fidx, hptr, wptr,
                                                      (float*)d_out, Q, F, D);
}
